// GCRNN_1889785610422
// MI455X (gfx1250) — matrix operand order repaired and checked by reading, not compiled, not run on hardware
//
#include <hip/hip_runtime.h>
#include <hip/hip_bf16.h>
#include <stdint.h>

#define TT 16
#define FF 128
#define DD 128

typedef __attribute__((ext_vector_type(16))) __bf16 v16bf;
typedef __attribute__((ext_vector_type(8)))  float  v8f;

union AFrag { v16bf v; __bf16 e[16]; };
union BFrag { v16bf v; uint4  q[2]; };

static __device__ __forceinline__ unsigned short f2bf(float f) {
  union { __bf16 b; unsigned short u; } cv;
  cv.b = (__bf16)f;
  return cv.u;
}

// Build a 16x32 bf16 A-fragment from a 128-float row in global memory.
// ISA layout (16-bit A 16x32): lanes 0-15 hold K {0..7, 16..23}, lanes 16-31
// hold K {8..15, 24..31}; elems 0..7 = 8 consecutive K at kbase, elems 8..15
// at kbase+16.
static __device__ __forceinline__ AFrag loadA(const float* __restrict__ row, int kbase) {
  AFrag a;
  float4 f0 = *(const float4*)(row + kbase);
  float4 f1 = *(const float4*)(row + kbase + 4);
  float4 f2 = *(const float4*)(row + kbase + 16);
  float4 f3 = *(const float4*)(row + kbase + 20);
  a.e[0]  = (__bf16)f0.x; a.e[1]  = (__bf16)f0.y; a.e[2]  = (__bf16)f0.z; a.e[3]  = (__bf16)f0.w;
  a.e[4]  = (__bf16)f1.x; a.e[5]  = (__bf16)f1.y; a.e[6]  = (__bf16)f1.z; a.e[7]  = (__bf16)f1.w;
  a.e[8]  = (__bf16)f2.x; a.e[9]  = (__bf16)f2.y; a.e[10] = (__bf16)f2.z; a.e[11] = (__bf16)f2.w;
  a.e[12] = (__bf16)f3.x; a.e[13] = (__bf16)f3.y; a.e[14] = (__bf16)f3.z; a.e[15] = (__bf16)f3.w;
  return a;
}

// ---------------- setup kernels ----------------

__global__ void count_deg_kernel(const int* __restrict__ ei, int E, int* __restrict__ cnt) {
  int e = blockIdx.x * blockDim.x + threadIdx.x;
  if (e < E) atomicAdd(&cnt[ei[E + e]], 1);   // dst = ei[E + e]
}

__global__ void dinv_kernel(const int* __restrict__ cnt, float* __restrict__ dinv, int n) {
  int i = blockIdx.x * blockDim.x + threadIdx.x;
  if (i < n) dinv[i] = rsqrtf((float)cnt[i] + 1.0f);   // +1 self-loop, always > 0
}

// Single-block exclusive scan (row_ptr[0]=0, row_ptr[i+1]=inclusive(cnt[0..i])).
__global__ __launch_bounds__(1024) void scan_kernel(const int* __restrict__ cnt,
                                                    int* __restrict__ row_ptr, int n) {
  __shared__ int sh[1024];
  __shared__ int carry;
  const int tid = threadIdx.x;
  if (tid == 0) carry = 0;
  __syncthreads();
  for (int base = 0; base < n; base += 1024) {
    int i = base + tid;
    int v = (i < n) ? cnt[i] : 0;
    sh[tid] = v;
    __syncthreads();
    for (int off = 1; off < 1024; off <<= 1) {
      int t = 0;
      if (tid >= off) t = sh[tid - off];
      __syncthreads();
      sh[tid] += t;
      __syncthreads();
    }
    int inc = sh[tid] + carry;
    if (i < n) row_ptr[i + 1] = inc;
    __syncthreads();
    if (tid == 1023) carry = inc;
    __syncthreads();
  }
  if (tid == 0) row_ptr[0] = 0;
}

__global__ void fill_csr_kernel(const int* __restrict__ ei, int E,
                                const int* __restrict__ row_ptr,
                                int* __restrict__ cursor,
                                const float* __restrict__ dinv,
                                int* __restrict__ col, float* __restrict__ ew) {
  int e = blockIdx.x * blockDim.x + threadIdx.x;
  if (e >= E) return;
  int s = ei[e];
  int d = ei[E + e];
  int pos = atomicAdd(&cursor[d], 1);
  int idx = row_ptr[d] + pos;
  col[idx] = s;
  ew[idx] = dinv[s] * dinv[d];
}

// Pre-swizzle the two 128x128 weights into WMMA B-fragment order (bf16).
// B (32x16, 16-bit): lanes 0-15 hold K {0..7, 16..23} of the tile, lanes 16-31 K {8..15, 24..31};
// element e (0..15) = K (kin & 7) + (kin & 16 ? 8 : 0). Fragment for (ktile,ntile) is 32 lanes x
// 16 bf16, stored lane-major so a lane reads 32 contiguous bytes.
__global__ void prep_w_kernel(const float* __restrict__ Wc, const float* __restrict__ Wl,
                              unsigned short* __restrict__ Wc_sw,
                              unsigned short* __restrict__ Wl_sw) {
  int idx = blockIdx.x * blockDim.x + threadIdx.x;
  if (idx >= 2 * 16384) return;
  int m = idx >> 14;
  int s = idx & 16383;
  int k  = s >> 7, nn  = s & 127;
  int kt = k >> 5, kin = k & 31;
  int nt = nn >> 4, nin = nn & 15;
  int lane = nin + ((kin & 8) ? 16 : 0);
  int e = (kin & 7) + ((kin & 16) ? 8 : 0);
  int dst = ((kt * 8 + nt) * 32 + lane) * 16 + e;
  float v = m ? Wl[s] : Wc[s];
  if (m) Wl_sw[dst] = f2bf(v); else Wc_sw[dst] = f2bf(v);
}

// ---------------- per-timestep kernels ----------------

// One block = 128 rows (8 waves x 16-row strip). Computes:
//   xw  = bf16( X[:,t,:] ) @ bf16(W_conv)                      -> ws
//   acc = (t>0 ? h_prev @ W_lin : 0) + b_conv + b_lin + xw*dinv^2  -> ws
__global__ __launch_bounds__(256) void step_gemm(
    const float* __restrict__ x, const float* __restrict__ hprev,
    const unsigned short* __restrict__ Wc_sw, const unsigned short* __restrict__ Wl_sw,
    const float* __restrict__ bc, const float* __restrict__ bl,
    const float* __restrict__ dinv,
    float* __restrict__ xw, float* __restrict__ acc,
    int n, int t, int first) {
  __shared__ unsigned short sW[16384];   // 32 KB: one swizzled weight matrix
  const int tid  = threadIdx.x;
  const int wave = tid >> 5;
  const int lane = tid & 31;
  const int half = lane >> 4;
  const int R    = blockIdx.x * 128 + wave * 16;

  // stage W_conv fragments into LDS
  {
    const uint4* g = (const uint4*)Wc_sw;
    uint4* s = (uint4*)sW;
    for (int i = tid; i < 2048; i += 256) s[i] = g[i];
  }
  __syncthreads();

  int ar  = R + (lane & 15);
  int arc = (ar < n) ? ar : (n - 1);           // clamp; OOB rows never stored
  const float* arow = x + ((size_t)arc * TT + t) * FF;

  v8f c[8];
  for (int i = 0; i < 8; i++)
    for (int j = 0; j < 8; j++) c[i][j] = 0.0f;

  // XW = X_t @ W_conv  (4 K-tiles x 8 N-tiles of 16x16x32 bf16 WMMA)
  for (int kt = 0; kt < 4; kt++) {
    AFrag a = loadA(arow, kt * 32 + half * 8);
    for (int nt = 0; nt < 8; nt++) {
      BFrag b;
      const uint4* p = (const uint4*)&sW[((kt * 8 + nt) * 32 + lane) * 16];
      b.q[0] = p[0];
      b.q[1] = p[1];
      c[nt] = __builtin_amdgcn_wmma_f32_16x16x32_bf16(
          false, a.v, false, b.v, (short)0, c[nt], false, false);
    }
  }

  // per-lane dinv^2 for its 8 output rows (C layout: lane<16 -> M=v, else M=v+8)
  float dv[8];
  for (int v = 0; v < 8; v++) {
    int r  = R + half * 8 + v;
    int rc = (r < n) ? r : (n - 1);
    float di = dinv[rc];
    dv[v] = di * di;
  }

  // store xw; fold self-loop + biases into accumulators as new C seed
  for (int nt = 0; nt < 8; nt++) {
    int colI = (lane & 15) + nt * 16;
    float bs = bc[colI] + bl[colI];
    for (int v = 0; v < 8; v++) {
      int r = R + half * 8 + v;
      if (r < n) xw[(size_t)r * DD + colI] = c[nt][v];
      c[nt][v] = bs + c[nt][v] * dv[v];
    }
  }

  // swap W_lin fragments into LDS
  __syncthreads();
  {
    const uint4* g = (const uint4*)Wl_sw;
    uint4* s = (uint4*)sW;
    for (int i = tid; i < 2048; i += 256) s[i] = g[i];
  }
  __syncthreads();

  if (!first) {  // uniform branch: EXEC stays all-ones around WMMA
    const float* hrow = hprev + (size_t)arc * DD;
    for (int kt = 0; kt < 4; kt++) {
      AFrag a = loadA(hrow, kt * 32 + half * 8);
      for (int nt = 0; nt < 8; nt++) {
        BFrag b;
        const uint4* p = (const uint4*)&sW[((kt * 8 + nt) * 32 + lane) * 16];
        b.q[0] = p[0];
        b.q[1] = p[1];
        c[nt] = __builtin_amdgcn_wmma_f32_16x16x32_bf16(
            false, a.v, false, b.v, (short)0, c[nt], false, false);
      }
    }
  }

  for (int nt = 0; nt < 8; nt++) {
    int colI = (lane & 15) + nt * 16;
    for (int v = 0; v < 8; v++) {
      int r = R + half * 8 + v;
      if (r < n) acc[(size_t)r * DD + colI] = c[nt][v];
    }
  }
}

// One wave per dst node; lane owns 4 features. Walks the CSR in-edge list
// (L2-resident xw rows), adds to the precomputed base, tanh, writes h and out.
__global__ __launch_bounds__(256) void step_gather(
    const float4* __restrict__ xw4, const float4* __restrict__ acc4,
    const int* __restrict__ row_ptr, const int* __restrict__ col,
    const float* __restrict__ ew,
    float4* __restrict__ h4, float4* __restrict__ out4, int n, int t) {
  const int wave = threadIdx.x >> 5;
  const int lane = threadIdx.x & 31;
  const int i = blockIdx.x * 8 + wave;      // wave-uniform
  if (i >= n) return;
  float4 a = acc4[(size_t)i * 32 + lane];
  const int beg = row_ptr[i];
  const int end = row_ptr[i + 1];
  for (int j = beg; j < end; j++) {
    int s  = col[j];
    float w = ew[j];
    float4 xs = xw4[(size_t)s * 32 + lane];
    a.x += w * xs.x;
    a.y += w * xs.y;
    a.z += w * xs.z;
    a.w += w * xs.w;
  }
  float4 hv;
  hv.x = tanhf(a.x);
  hv.y = tanhf(a.y);
  hv.z = tanhf(a.z);
  hv.w = tanhf(a.w);
  h4[(size_t)i * 32 + lane] = hv;
  out4[((size_t)i * TT + t) * 32 + lane] = hv;   // out[i, t, :]
}

// ---------------- launcher ----------------

extern "C" void kernel_launch(void* const* d_in, const int* in_sizes, int n_in,
                              void* d_out, int out_size, void* d_ws, size_t ws_size,
                              hipStream_t stream) {
  const float* x  = (const float*)d_in[0];
  const int*   ei = (const int*)d_in[1];
  const float* Wc = (const float*)d_in[2];
  const float* bc = (const float*)d_in[3];
  const float* Wl = (const float*)d_in[4];
  const float* bl = (const float*)d_in[5];
  float* out = (float*)d_out;

  const int N = in_sizes[0] / (TT * FF);
  const int E = in_sizes[1] / 2;

  // workspace carve (256B aligned)
  char* p = (char*)d_ws;
  auto carve = [&](size_t bytes) -> void* {
    void* r = (void*)p;
    p += (bytes + 255) & ~(size_t)255;
    return r;
  };
  int*            cnt     = (int*)carve((size_t)N * 4);
  float*          dinv    = (float*)carve((size_t)N * 4);
  int*            row_ptr = (int*)carve((size_t)(N + 1) * 4);
  int*            cursor  = (int*)carve((size_t)N * 4);
  int*            col     = (int*)carve((size_t)E * 4);
  float*          ewt     = (float*)carve((size_t)E * 4);
  unsigned short* Wc_sw   = (unsigned short*)carve(16384 * 2);
  unsigned short* Wl_sw   = (unsigned short*)carve(16384 * 2);
  float*          xw      = (float*)carve((size_t)N * DD * 4);
  float*          acc     = (float*)carve((size_t)N * DD * 4);
  float*          hbuf    = (float*)carve((size_t)N * DD * 4);
  (void)ws_size; (void)n_in; (void)out_size;

  hipMemsetAsync(cnt, 0, (size_t)N * 4, stream);
  hipMemsetAsync(cursor, 0, (size_t)N * 4, stream);

  count_deg_kernel<<<(E + 255) / 256, 256, 0, stream>>>(ei, E, cnt);
  dinv_kernel<<<(N + 255) / 256, 256, 0, stream>>>(cnt, dinv, N);
  scan_kernel<<<1, 1024, 0, stream>>>(cnt, row_ptr, N);
  fill_csr_kernel<<<(E + 255) / 256, 256, 0, stream>>>(ei, E, row_ptr, cursor, dinv, col, ewt);
  prep_w_kernel<<<(2 * 16384 + 255) / 256, 256, 0, stream>>>(Wc, Wl, Wc_sw, Wl_sw);

  const int gemm_blocks   = (N + 127) / 128;
  const int gather_blocks = (N + 7) / 8;
  for (int t = 0; t < TT; t++) {
    step_gemm<<<gemm_blocks, 256, 0, stream>>>(x, hbuf, Wc_sw, Wl_sw, bc, bl, dinv,
                                               xw, acc, N, t, t == 0 ? 1 : 0);
    step_gather<<<gather_blocks, 256, 0, stream>>>((const float4*)xw, (const float4*)acc,
                                                   row_ptr, col, ewt,
                                                   (float4*)hbuf, (float4*)out, N, t);
  }
}
